// MultiHeadODELinear_5463198400842
// MI455X (gfx1250) — matrix operand order repaired and checked by reading, not compiled, not run on hardware
//
#include <hip/hip_runtime.h>

typedef __attribute__((ext_vector_type(16))) _Float16 v16h;
typedef __attribute__((ext_vector_type(8)))  _Float16 v8h;
typedef __attribute__((ext_vector_type(8)))  float    v8f;
typedef int v4i __attribute__((vector_size(16)));

#define GLOBAL_AS __attribute__((address_space(1)))
#define LDS_AS    __attribute__((address_space(3)))

#define D_IN      1024
#define D_OUT     1024
#define NUM_HEADS 16
#define HEAD_DIM  64
#define ORDERS    4

#if __has_builtin(__builtin_amdgcn_global_load_async_to_lds_b128) && \
    __has_builtin(__builtin_amdgcn_s_wait_asynccnt)
#define USE_ASYNC 1
#else
#define USE_ASYNC 0
#endif

// =====================================================================
// Kernel 1: per-head truncated matrix exponential  M_h = sum (tA)^k/k!
// plus effective bias b_eff = M_h @ b_h.  16 blocks x 64 threads.
// =====================================================================
__global__ __launch_bounds__(64) void expm_kernel(const float* __restrict__ A,
                                                  const float* __restrict__ bvec,
                                                  const float* __restrict__ tptr,
                                                  float* __restrict__ Ms,
                                                  float* __restrict__ beff) {
    __shared__ float sA[HEAD_DIM * HEAD_DIM];
    __shared__ float sT[2][HEAD_DIM * HEAD_DIM];

    const int h = blockIdx.x;
    const int i = threadIdx.x;           // row owned by this thread
    const float t = *tptr;
    const float* Ah = A + (size_t)h * HEAD_DIM * HEAD_DIM;

    for (int idx = i; idx < HEAD_DIM * HEAD_DIM; idx += 64) sA[idx] = Ah[idx];
    __syncthreads();

    float Mrow[HEAD_DIM];
    // order 1: T1 = t*A ; M = I + T1
    #pragma unroll
    for (int j = 0; j < HEAD_DIM; ++j) {
        float v = t * sA[i * HEAD_DIM + j];
        sT[0][i * HEAD_DIM + j] = v;
        Mrow[j] = ((i == j) ? 1.0f : 0.0f) + v;
    }
    __syncthreads();

    int buf = 0;
    for (int k = 2; k <= ORDERS; ++k) {
        const float scale = t / (float)k;
        #pragma unroll
        for (int j = 0; j < HEAD_DIM; ++j) {
            float s = 0.0f;
            #pragma unroll 4
            for (int l = 0; l < HEAD_DIM; ++l)
                s += sA[i * HEAD_DIM + l] * sT[buf][l * HEAD_DIM + j];
            s *= scale;
            sT[buf ^ 1][i * HEAD_DIM + j] = s;
            Mrow[j] += s;
        }
        buf ^= 1;
        __syncthreads();
    }

    float* Mout = Ms + ((size_t)h * HEAD_DIM + i) * HEAD_DIM;
    #pragma unroll
    for (int j = 0; j < HEAD_DIM; ++j) Mout[j] = Mrow[j];

    float s = 0.0f;
    #pragma unroll
    for (int j = 0; j < HEAD_DIM; ++j) s += Mrow[j] * bvec[h * HEAD_DIM + j];
    beff[h * HEAD_DIM + i] = s;
}

// =====================================================================
// Kernel 2: W_eff[n,k] = sum_j M_h[n',j] * W[h*64+j, k], stored as two
// f16 planes (hi + residual lo) for split-precision WMMA.
// =====================================================================
__global__ __launch_bounds__(256) void weff_kernel(const float* __restrict__ Ms,
                                                   const float* __restrict__ W,
                                                   _Float16* __restrict__ Wh,
                                                   _Float16* __restrict__ Wl) {
    const int k = blockIdx.x * 256 + threadIdx.x;   // 0..1023
    const int n = blockIdx.y;                       // 0..1023
    const int h = n >> 6;
    const int nn = n & 63;
    const float* Mrow = Ms + ((size_t)h * HEAD_DIM + nn) * HEAD_DIM;

    float s = 0.0f;
    #pragma unroll 8
    for (int j = 0; j < HEAD_DIM; ++j)
        s += Mrow[j] * W[(size_t)(h * HEAD_DIM + j) * D_IN + k];

    _Float16 hi = (_Float16)s;
    _Float16 lo = (_Float16)(s - (float)hi);
    Wh[(size_t)n * D_IN + k] = hi;
    Wl[(size_t)n * D_IN + k] = lo;
}

// =====================================================================
// Kernel 3: out[M,1024] = X[M,1024] @ W_eff^T + b_eff via f16 WMMA with
// split-precision (x_hi*W_hi + x_lo*W_hi + x_hi*W_lo), fp32 accumulate.
// Block tile 128x128, 8 waves each owning a 32x64 tile.  X tile is
// staged raw (f32) into double-buffered LDS via async-to-LDS DMA.
// =====================================================================
#define BM 128
#define BN 128
#define BK 32
#define LDSF 36   // f32 per LDS row: 36*m mod 64 is a permutation -> conflict-free b128 reads

__global__ __launch_bounds__(256) void ode_gemm_kernel(const float* __restrict__ X,
                                                       const _Float16* __restrict__ Wh,
                                                       const _Float16* __restrict__ Wl,
                                                       const float* __restrict__ beff,
                                                       float* __restrict__ out) {
#if USE_ASYNC
    __shared__ float sX[2][BM * LDSF];
#else
    __shared__ float sX[1][BM * LDSF];
#endif

    const int tid   = threadIdx.x;
    const int lane  = tid & 31;
    const int wave  = tid >> 5;
    const int wm    = wave & 3;     // 4 waves along M
    const int wn    = wave >> 2;    // 2 waves along N
    const int l16   = lane & 15;
    const int khalf = lane >> 4;    // 0/1 : which K-half this lane holds

    const int mBase = blockIdx.x * BM;
    const int nBase = blockIdx.y * BN;

    v8f acc[2][4];
    #pragma unroll
    for (int mi = 0; mi < 2; ++mi)
        #pragma unroll
        for (int ni = 0; ni < 4; ++ni)
            acc[mi][ni] = v8f{0.f, 0.f, 0.f, 0.f, 0.f, 0.f, 0.f, 0.f};

#if USE_ASYNC
    // Each thread DMA-copies 4 x 16B chunks of the 128x32 f32 tile.
    auto stage_async = [&](int kb, int bufi) {
        #pragma unroll
        for (int v = 0; v < 4; ++v) {
            const int c   = (tid << 2) + v;     // chunk id 0..1023
            const int row = c >> 3;             // 0..127
            const int kc  = (c & 7) << 2;       // 0,4,..,28 (f32)
            const float* g = X + (size_t)(mBase + row) * D_IN + kb + kc;
            float* l = &sX[bufi][row * LDSF + kc];
            __builtin_amdgcn_global_load_async_to_lds_b128(
                (GLOBAL_AS v4i*)g,
                (LDS_AS v4i*)l,
                0, 0);
        }
    };
    stage_async(0, 0);
#endif

    for (int kb = 0, it = 0; kb < D_IN; kb += BK, ++it) {
#if USE_ASYNC
        const int cur = it & 1;
        __builtin_amdgcn_s_wait_asynccnt(0);   // own portion of tile `it` landed
        __syncthreads();                        // whole tile landed; prev reads done
        if (kb + BK < D_IN) stage_async(kb + BK, cur ^ 1);
#else
        const int cur = 0;
        __syncthreads();
        #pragma unroll
        for (int v = 0; v < 4; ++v) {
            const int c   = (tid << 2) + v;
            const int row = c >> 3;
            const int kc  = (c & 7) << 2;
            float4 f = *(const float4*)(X + (size_t)(mBase + row) * D_IN + kb + kc);
            *(float4*)&sX[0][row * LDSF + kc] = f;
        }
        __syncthreads();
#endif

        // --- A fragments: read f32 from LDS, split to f16 hi/lo on the fly ---
        v16h ah[2], al[2];
        #pragma unroll
        for (int mi = 0; mi < 2; ++mi) {
            const int m = wm * 32 + mi * 16 + l16;
            const float* pa = &sX[cur][m * LDSF + khalf * 16];
            float4 f0 = *(const float4*)(pa);
            float4 f1 = *(const float4*)(pa + 4);
            float4 f2 = *(const float4*)(pa + 8);
            float4 f3 = *(const float4*)(pa + 12);
            float xs[16] = {f0.x, f0.y, f0.z, f0.w, f1.x, f1.y, f1.z, f1.w,
                            f2.x, f2.y, f2.z, f2.w, f3.x, f3.y, f3.z, f3.w};
            v16h H, L;
            #pragma unroll
            for (int j = 0; j < 16; ++j) {
                _Float16 hh = (_Float16)xs[j];
                H[j] = hh;
                L[j] = (_Float16)(xs[j] - (float)hh);
            }
            ah[mi] = H;
            al[mi] = L;
        }

        // --- B fragments (contiguous k order, same as A): b[j] = Weff[n, kb+khalf*16+j]
        v16h bhv[4], blv[4];
        #pragma unroll
        for (int ni = 0; ni < 4; ++ni) {
            const int n = nBase + wn * 64 + ni * 16 + l16;
            const size_t off = (size_t)n * D_IN + kb + khalf * 16;
            bhv[ni] = *(const v16h*)(Wh + off);
            blv[ni] = *(const v16h*)(Wl + off);
        }

        #pragma unroll
        for (int mi = 0; mi < 2; ++mi)
            #pragma unroll
            for (int ni = 0; ni < 4; ++ni) {
                acc[mi][ni] = __builtin_amdgcn_wmma_f32_16x16x32_f16(
                    false, ah[mi], false, bhv[ni], (short)0, acc[mi][ni], false, false);
                acc[mi][ni] = __builtin_amdgcn_wmma_f32_16x16x32_f16(
                    false, al[mi], false, bhv[ni], (short)0, acc[mi][ni], false, false);
                acc[mi][ni] = __builtin_amdgcn_wmma_f32_16x16x32_f16(
                    false, ah[mi], false, blv[ni], (short)0, acc[mi][ni], false, false);
            }
    }

    // --- epilogue: C 16x16 f32 layout; add bias and store ---
    #pragma unroll
    for (int mi = 0; mi < 2; ++mi) {
        const int rbase = mBase + wm * 32 + mi * 16 + khalf * 8;
        #pragma unroll
        for (int ni = 0; ni < 4; ++ni) {
            const int n = nBase + wn * 64 + ni * 16 + l16;
            const float bias = beff[n];
            #pragma unroll
            for (int e = 0; e < 8; ++e)
                out[(size_t)(rbase + e) * D_OUT + n] = acc[mi][ni][e] + bias;
        }
    }
}

// =====================================================================
// Workspace layout (bytes):
//   [0,       262144)  M_h  fp32  16*64*64
//   [262144,  266240)  b_eff fp32 1024
//   [266240,  2363392) W_eff hi plane, f16, 1024*1024
//   [2363392, 4460544) W_eff lo plane, f16, 1024*1024
// =====================================================================
extern "C" void kernel_launch(void* const* d_in, const int* in_sizes, int n_in,
                              void* d_out, int out_size, void* d_ws, size_t ws_size,
                              hipStream_t stream) {
    const float* X  = (const float*)d_in[0];   // [B,S,1024] fp32
    const float* t  = (const float*)d_in[1];   // scalar
    const float* W  = (const float*)d_in[2];   // [1024,1024]
    const float* b  = (const float*)d_in[3];   // [1024]
    const float* A  = (const float*)d_in[4];   // [16,64,64]
    float* out      = (float*)d_out;

    char* ws = (char*)d_ws;
    float*    Ms   = (float*)ws;
    float*    beff = (float*)(ws + 262144);
    _Float16* Wh   = (_Float16*)(ws + 266240);
    _Float16* Wl   = (_Float16*)(ws + 266240 + 2097152);

    const int M = in_sizes[0] / D_IN;          // 32768 rows

    expm_kernel<<<NUM_HEADS, 64, 0, stream>>>(A, b, t, Ms, beff);
    weff_kernel<<<dim3(D_IN / 256, D_OUT), 256, 0, stream>>>(Ms, W, Wh, Wl);
    ode_gemm_kernel<<<dim3(M / BM, D_OUT / BN), 256, 0, stream>>>(X, Wh, Wl, beff, out);
}
